// Transformer_21603685499665
// MI455X (gfx1250) — compile-verified, matrix operand order checked by reading, not run on hardware
//
#include <hip/hip_runtime.h>
#include <hip/hip_bf16.h>
#include <stdint.h>

#define D_    1024
#define HD_   64
#define NH_   16
#define NKV_  4
#define HID_  2048
#define L_    2
#define E_    8
#define V_    32000
#define B_    2
#define S_    512
#define T_    (B_*S_)
#define EPS_  1e-5f

// Async global->LDS staging (ASYNCcnt path). Set to 0 to fall back to plain
// loads+ds_stores if the assembler rejects the mnemonic.
#define USE_ASYNC_LDS 1

typedef __attribute__((ext_vector_type(16))) _Float16 v16h;
typedef __attribute__((ext_vector_type(8)))  float    v8f;

// ---------------------------------------------------------------------------
// Load one WMMA f16 A/B fragment from an fp32 row (row-major, K contiguous).
// 16-bit A 16x32 layout (ISA 7.12.2): lane L: M=L%16, halfSel=L/16;
//   VGPR0-3 hold K = halfSel*8 + 0..7
//   VGPR4-7 hold K = 16 + halfSel*8 + 0..7
// B (32x16) is symmetric: lane reads row N=L%16 of W (since B[k][n]=W[n][k]).
// Works for both global and LDS (generic) pointers.
// ---------------------------------------------------------------------------
static __device__ __forceinline__ v16h load_frag_f32(const float* __restrict__ rowp,
                                                     int k0, int halfSel) {
  const float4 f0 = *(const float4*)(rowp + k0 + halfSel * 8);
  const float4 f1 = *(const float4*)(rowp + k0 + halfSel * 8 + 4);
  const float4 f2 = *(const float4*)(rowp + k0 + 16 + halfSel * 8);
  const float4 f3 = *(const float4*)(rowp + k0 + 16 + halfSel * 8 + 4);
  v16h r;
  r[0]  = (_Float16)f0.x; r[1]  = (_Float16)f0.y; r[2]  = (_Float16)f0.z; r[3]  = (_Float16)f0.w;
  r[4]  = (_Float16)f1.x; r[5]  = (_Float16)f1.y; r[6]  = (_Float16)f1.z; r[7]  = (_Float16)f1.w;
  r[8]  = (_Float16)f2.x; r[9]  = (_Float16)f2.y; r[10] = (_Float16)f2.z; r[11] = (_Float16)f2.w;
  r[12] = (_Float16)f3.x; r[13] = (_Float16)f3.y; r[14] = (_Float16)f3.z; r[15] = (_Float16)f3.w;
  return r;
}

static __device__ __forceinline__ v8f v8f_zero() {
  v8f z = {0.f, 0.f, 0.f, 0.f, 0.f, 0.f, 0.f, 0.f};
  return z;
}

#if USE_ASYNC_LDS
// Copy 64 contiguous bytes global->LDS with 4 async b128 transfers.
// INST_OFFSET is added to BOTH the LDS and global address (ISA 10.x), so one
// (lds, global) pair + offsets walks both sides in lockstep.
static __device__ __forceinline__ void async_cp64(uint32_t lds_off, const float* g) {
  const uint64_t ga = (uint64_t)(uintptr_t)g;
  asm volatile("global_load_async_to_lds_b128 %0, %1, off"           :: "v"(lds_off), "v"(ga) : "memory");
  asm volatile("global_load_async_to_lds_b128 %0, %1, off offset:16" :: "v"(lds_off), "v"(ga) : "memory");
  asm volatile("global_load_async_to_lds_b128 %0, %1, off offset:32" :: "v"(lds_off), "v"(ga) : "memory");
  asm volatile("global_load_async_to_lds_b128 %0, %1, off offset:48" :: "v"(lds_off), "v"(ga) : "memory");
}
#endif

// ---------------------------------------------------------------------------
// Multi-wave GEMM with LDS-staged A tile (for N >= 512 call sites):
//   C[M,N] = A[M,K] * W[N,K]^T   (fp32 memory, f16 WMMA, f32 accumulate)
// Block = 256 threads = 8 waves. All waves share a 64-row A tile, double
// buffered in LDS and staged with async global->LDS copies (ASYNCcnt).
// Wave w computes a 64x64 sub-tile at columns blockN0 + w*64 (4x4 WMMA tiles).
// K % 64 == 0, M % 64 == 0. Partial N handled by wave-uniform 'active' guard
// (inactive waves still stage + hit barriers, so EXEC is full at every WMMA).
// ---------------------------------------------------------------------------
#define LDS_PITCH 68   // 64 + 4 floats pad: de-conflicts LDS banks on frag reads

template <bool ACCUM>
__global__ __launch_bounds__(256)
void gemm_lds_kernel(const float* __restrict__ A,
                     const float* __restrict__ W,
                     float* __restrict__ C,
                     int M, int N, int K) {
  const int tid  = threadIdx.x;
  const int wave = tid >> 5;
  const int lane = tid & 31;
  const int lm   = lane & 15;
  const int hs   = lane >> 4;
  const int m0   = blockIdx.x * 64;
  const int n0   = blockIdx.y * 512 + wave * 64;
  const bool active = (n0 + 63 < N) || (n0 < N);   // wave-uniform

  __shared__ float Abuf[2][64 * LDS_PITCH];

  // Staging map: 4 threads per row, 16 floats (64B) per thread per chunk.
  const int srow = tid >> 2;
  const int sseg = tid & 3;
  const float* gsrc = A + (size_t)(m0 + srow) * K + sseg * 16;
  const uint32_t lds_off[2] = {
    (uint32_t)(uintptr_t)&Abuf[0][srow * LDS_PITCH + sseg * 16],
    (uint32_t)(uintptr_t)&Abuf[1][srow * LDS_PITCH + sseg * 16]
  };

  const int n0c = active ? n0 : 0;
  const float* wrow[4];
#pragma unroll
  for (int n = 0; n < 4; ++n) wrow[n] = W + (size_t)(n0c + n * 16 + lm) * K;

  v8f c[4][4];
#pragma unroll
  for (int m = 0; m < 4; ++m)
#pragma unroll
    for (int n = 0; n < 4; ++n) {
      if (ACCUM) {
        if (active) {
#pragma unroll
          for (int r = 0; r < 8; ++r)
            c[m][n][r] = C[(size_t)(m0 + m * 16 + hs * 8 + r) * N + (n0 + n * 16 + lm)];
        } else {
          c[m][n] = v8f_zero();
        }
      } else {
        c[m][n] = v8f_zero();
      }
    }

  const int nch = K / 64;

#if USE_ASYNC_LDS
  async_cp64(lds_off[0], gsrc);                       // preload chunk 0
#else
  {
    const float4* s = (const float4*)gsrc;
    float4* d = (float4*)((char*)&Abuf[0][srow * LDS_PITCH + sseg * 16]);
    d[0] = s[0]; d[1] = s[1]; d[2] = s[2]; d[3] = s[3];
  }
#endif

  for (int ch = 0; ch < nch; ++ch) {
    const int cur = ch & 1;
    if (ch + 1 < nch) {
#if USE_ASYNC_LDS
      async_cp64(lds_off[cur ^ 1], gsrc + (ch + 1) * 64);
      asm volatile("s_wait_asynccnt 4" ::: "memory"); // chunk ch done; ch+1 in flight
#else
      const float4* s = (const float4*)(gsrc + (ch + 1) * 64);
      float4* d = (float4*)((char*)&Abuf[cur ^ 1][srow * LDS_PITCH + sseg * 16]);
      d[0] = s[0]; d[1] = s[1]; d[2] = s[2]; d[3] = s[3];
#endif
    } else {
#if USE_ASYNC_LDS
      asm volatile("s_wait_asynccnt 0" ::: "memory");
#endif
    }
    __syncthreads();                                   // A tile visible block-wide

    if (active) {
#pragma unroll
      for (int kk = 0; kk < 64; kk += 32) {
        v16h a[4];
#pragma unroll
        for (int m = 0; m < 4; ++m)
          a[m] = load_frag_f32(&Abuf[cur][(m * 16 + lm) * LDS_PITCH], kk, hs);
#pragma unroll
        for (int n = 0; n < 4; ++n) {
          v16h b = load_frag_f32(wrow[n], ch * 64 + kk, hs);
          __builtin_prefetch(wrow[n] + ch * 64 + kk + 128, 0, 0);
#pragma unroll
          for (int m = 0; m < 4; ++m)
            c[m][n] = __builtin_amdgcn_wmma_f32_16x16x32_f16(false, a[m], false, b,
                                                             (short)0, c[m][n], false, false);
        }
      }
    }
    __syncthreads();                                   // before buf 'cur' is re-staged
  }

  if (active) {
#pragma unroll
    for (int m = 0; m < 4; ++m)
#pragma unroll
      for (int n = 0; n < 4; ++n)
#pragma unroll
        for (int r = 0; r < 8; ++r)
          C[(size_t)(m0 + m * 16 + hs * 8 + r) * N + (n0 + n * 16 + lm)] = c[m][n][r];
  }
}

// ---------------------------------------------------------------------------
// Single-wave register-blocked GEMM (used for narrow N and as MoE base).
// ---------------------------------------------------------------------------
template <int MT, int NT, bool ACCUM>
__global__ void gemm_wmma_kernel(const float* __restrict__ A,
                                 const float* __restrict__ W,
                                 float* __restrict__ C,
                                 int M, int N, int K) {
  const int lane = threadIdx.x & 31;
  const int lm = lane & 15;
  const int hs = lane >> 4;
  const int m0 = blockIdx.x * (16 * MT);
  const int n0 = blockIdx.y * (16 * NT);

  const float* arow[MT];
#pragma unroll
  for (int m = 0; m < MT; ++m) arow[m] = A + (size_t)(m0 + m * 16 + lm) * K;
  const float* wrow[NT];
#pragma unroll
  for (int n = 0; n < NT; ++n) wrow[n] = W + (size_t)(n0 + n * 16 + lm) * K;

  v8f c[MT][NT];
#pragma unroll
  for (int m = 0; m < MT; ++m)
#pragma unroll
    for (int n = 0; n < NT; ++n) {
      if (ACCUM) {
#pragma unroll
        for (int r = 0; r < 8; ++r)
          c[m][n][r] = C[(size_t)(m0 + m * 16 + hs * 8 + r) * N + (n0 + n * 16 + lm)];
      } else {
        c[m][n] = v8f_zero();
      }
    }

  for (int k0 = 0; k0 < K; k0 += 32) {
    v16h a[MT];
#pragma unroll
    for (int m = 0; m < MT; ++m) {
      a[m] = load_frag_f32(arow[m], k0, hs);
      __builtin_prefetch(arow[m] + k0 + 64, 0, 0);
    }
#pragma unroll
    for (int n = 0; n < NT; ++n) {
      v16h b = load_frag_f32(wrow[n], k0, hs);
      __builtin_prefetch(wrow[n] + k0 + 64, 0, 0);
#pragma unroll
      for (int m = 0; m < MT; ++m)
        c[m][n] = __builtin_amdgcn_wmma_f32_16x16x32_f16(false, a[m], false, b,
                                                         (short)0, c[m][n], false, false);
    }
  }

#pragma unroll
  for (int m = 0; m < MT; ++m)
#pragma unroll
    for (int n = 0; n < NT; ++n)
#pragma unroll
      for (int r = 0; r < 8; ++r)
        C[(size_t)(m0 + m * 16 + hs * 8 + r) * N + (n0 + n * 16 + lm)] = c[m][n][r];
}

// ---------------------------------------------------------------------------
// MoE fused W1/W3 GEMM over gathered rows:
//   G[r, :] = silu(x[tok[r]] @ W1^T) * (x[tok[r]] @ W3^T)   for r < count
// ---------------------------------------------------------------------------
__global__ void moe_gemm13_kernel(const float* __restrict__ xs,
                                  const float* __restrict__ W1,
                                  const float* __restrict__ W3,
                                  const int* __restrict__ list,
                                  const int* __restrict__ cnts, int e,
                                  float* __restrict__ G) {
  const int count = cnts[e];
  const int m0 = blockIdx.x * 32;
  if (m0 >= count) return;             // uniform per wave -> safe early exit
  const int lane = threadIdx.x & 31;
  const int lm = lane & 15;
  const int hs = lane >> 4;
  const int n0 = blockIdx.y * 32;

  const float* arow[2];
#pragma unroll
  for (int m = 0; m < 2; ++m) {
    int gr = m0 + m * 16 + lm;
    if (gr >= count) gr = count - 1;   // clamp: keep EXEC full for WMMA
    arow[m] = xs + (size_t)list[gr] * D_;
  }
  const float* w1row[2];
  const float* w3row[2];
#pragma unroll
  for (int n = 0; n < 2; ++n) {
    w1row[n] = W1 + (size_t)(n0 + n * 16 + lm) * D_;
    w3row[n] = W3 + (size_t)(n0 + n * 16 + lm) * D_;
  }

  v8f c1[2][2], c3[2][2];
#pragma unroll
  for (int m = 0; m < 2; ++m)
#pragma unroll
    for (int n = 0; n < 2; ++n) { c1[m][n] = v8f_zero(); c3[m][n] = v8f_zero(); }

  for (int k0 = 0; k0 < D_; k0 += 32) {
    v16h a[2];
#pragma unroll
    for (int m = 0; m < 2; ++m) a[m] = load_frag_f32(arow[m], k0, hs);
#pragma unroll
    for (int n = 0; n < 2; ++n) {
      v16h b1 = load_frag_f32(w1row[n], k0, hs);
      v16h b3 = load_frag_f32(w3row[n], k0, hs);
#pragma unroll
      for (int m = 0; m < 2; ++m) {
        c1[m][n] = __builtin_amdgcn_wmma_f32_16x16x32_f16(false, a[m], false, b1,
                                                          (short)0, c1[m][n], false, false);
        c3[m][n] = __builtin_amdgcn_wmma_f32_16x16x32_f16(false, a[m], false, b3,
                                                          (short)0, c3[m][n], false, false);
      }
    }
  }

#pragma unroll
  for (int m = 0; m < 2; ++m)
#pragma unroll
    for (int n = 0; n < 2; ++n)
#pragma unroll
      for (int r = 0; r < 8; ++r) {
        const int rp = m0 + m * 16 + hs * 8 + r;
        if (rp < count) {
          const float x1 = c1[m][n][r];
          const float sl = x1 / (1.f + __expf(-x1));     // silu
          G[(size_t)rp * HID_ + (n0 + n * 16 + lm)] = sl * c3[m][n][r];
        }
      }
}

// ---------------------------------------------------------------------------
// MoE W2 GEMM + weighted scatter-add:  h[tok[r]] += w[r] * (G[r] @ W2^T)
// ---------------------------------------------------------------------------
__global__ void moe_gemm2_kernel(const float* __restrict__ G,
                                 const float* __restrict__ W2,
                                 const int* __restrict__ list,
                                 const float* __restrict__ wlist,
                                 const int* __restrict__ cnts, int e,
                                 float* __restrict__ h) {
  const int count = cnts[e];
  const int m0 = blockIdx.x * 32;
  if (m0 >= count) return;
  const int lane = threadIdx.x & 31;
  const int lm = lane & 15;
  const int hs = lane >> 4;
  const int n0 = blockIdx.y * 64;

  const float* arow[2];
#pragma unroll
  for (int m = 0; m < 2; ++m) {
    int gr = m0 + m * 16 + lm;
    if (gr >= count) gr = count - 1;
    arow[m] = G + (size_t)gr * HID_;
  }
  const float* wrow[4];
#pragma unroll
  for (int n = 0; n < 4; ++n) wrow[n] = W2 + (size_t)(n0 + n * 16 + lm) * HID_;

  v8f c[2][4];
#pragma unroll
  for (int m = 0; m < 2; ++m)
#pragma unroll
    for (int n = 0; n < 4; ++n) c[m][n] = v8f_zero();

  for (int k0 = 0; k0 < HID_; k0 += 32) {
    v16h a[2];
#pragma unroll
    for (int m = 0; m < 2; ++m) a[m] = load_frag_f32(arow[m], k0, hs);
#pragma unroll
    for (int n = 0; n < 4; ++n) {
      v16h b = load_frag_f32(wrow[n], k0, hs);
      __builtin_prefetch(wrow[n] + k0 + 64, 0, 0);
#pragma unroll
      for (int m = 0; m < 2; ++m)
        c[m][n] = __builtin_amdgcn_wmma_f32_16x16x32_f16(false, a[m], false, b,
                                                         (short)0, c[m][n], false, false);
    }
  }

#pragma unroll
  for (int m = 0; m < 2; ++m)
#pragma unroll
    for (int n = 0; n < 4; ++n)
#pragma unroll
      for (int r = 0; r < 8; ++r) {
        const int rp = m0 + m * 16 + hs * 8 + r;
        if (rp < count) {
          const int tok = list[rp];
          const float wgt = wlist[rp];
          h[(size_t)tok * D_ + (n0 + n * 16 + lm)] += wgt * c[m][n][r];
        }
      }
}

// ---------------------------------------------------------------------------
// Elementwise / small kernels
// ---------------------------------------------------------------------------
__global__ void embed_kernel(const int* __restrict__ ids,
                             const float* __restrict__ emb,
                             float* __restrict__ h) {
  const int t = blockIdx.x;
  const int id = ids[t];
  for (int d = threadIdx.x; d < D_; d += 256)
    h[(size_t)t * D_ + d] = emb[(size_t)id * D_ + d];
}

__global__ void rmsnorm_kernel(const float* __restrict__ x,
                               const float* __restrict__ w,
                               float* __restrict__ o) {
  const int t = blockIdx.x;
  __shared__ float red[256];
  float s = 0.f;
  for (int d = threadIdx.x; d < D_; d += 256) {
    const float v = x[(size_t)t * D_ + d];
    s += v * v;
  }
  red[threadIdx.x] = s;
  __syncthreads();
  for (int off = 128; off > 0; off >>= 1) {
    if (threadIdx.x < off) red[threadIdx.x] += red[threadIdx.x + off];
    __syncthreads();
  }
  const float inv = rsqrtf(red[0] / (float)D_ + EPS_);
  for (int d = threadIdx.x; d < D_; d += 256)
    o[(size_t)t * D_ + d] = x[(size_t)t * D_ + d] * inv * w[d];
}

__global__ void rope_kernel(float* __restrict__ x, int nheads) {
  const int t = blockIdx.x;
  const int s = t % S_;                 // positions = arange(S), tokens batch-major
  const int idx = threadIdx.x;          // nheads*32 threads, one per rotation pair
  const int hh = idx >> 5;
  const int p = idx & 31;
  const float freq = powf(10000.f, -((float)(2 * p) / (float)HD_));
  const float ang = (float)s * freq;
  const float cs = cosf(ang), sn = sinf(ang);
  float* base = x + (size_t)t * nheads * HD_ + hh * HD_ + 2 * p;
  const float r = base[0], i = base[1];
  base[0] = r * cs - i * sn;
  base[1] = r * sn + i * cs;
}

// One wave per (batch, head, query): causal online-softmax attention.
__global__ void attention_kernel(const float* __restrict__ q,
                                 const float* __restrict__ k,
                                 const float* __restrict__ v,
                                 float* __restrict__ o) {
  const int qpos = blockIdx.x;
  const int hh = blockIdx.y;
  const int b = blockIdx.z;
  const int lane = threadIdx.x;         // 32 lanes, 2 head-dims each
  const int t = b * S_ + qpos;
  const int kvh = hh / (NH_ / NKV_);
  const float2 qv = *(const float2*)(q + (size_t)t * (NH_ * HD_) + hh * HD_ + 2 * lane);
  float m = -3.0e38f, l = 0.f, a0 = 0.f, a1 = 0.f;
  const float scale = 0.125f;           // HD^-0.5
  for (int kk = 0; kk <= qpos; ++kk) {
    const size_t off = (size_t)(b * S_ + kk) * (NKV_ * HD_) + kvh * HD_ + 2 * lane;
    const float2 kv2 = *(const float2*)(k + off);
    float p = qv.x * kv2.x + qv.y * kv2.y;
    for (int sh = 16; sh > 0; sh >>= 1) p += __shfl_xor(p, sh, 32);
    p *= scale;
    const float mn = fmaxf(m, p);
    const float corr = __expf(m - mn);
    const float wgt = __expf(p - mn);
    const float2 vv = *(const float2*)(v + off);
    l = l * corr + wgt;
    a0 = a0 * corr + wgt * vv.x;
    a1 = a1 * corr + wgt * vv.y;
    m = mn;
  }
  const float inv = 1.f / l;
  float* op = o + (size_t)t * (NH_ * HD_) + hh * HD_ + 2 * lane;
  op[0] = a0 * inv;
  op[1] = a1 * inv;
}

__global__ void zero_cnts_kernel(int* cnts) {
  if (threadIdx.x < E_) cnts[threadIdx.x] = 0;
}

// One wave per token: gate logits (D x E), top-2 + softmax, append to
// per-expert token lists.
__global__ void moe_gate_kernel(const float* __restrict__ xs,
                                const float* __restrict__ gw,
                                int* __restrict__ cnts,
                                int* __restrict__ lists,
                                float* __restrict__ wlists) {
  const int t = blockIdx.x;
  const int lane = threadIdx.x;
  float acc[E_];
#pragma unroll
  for (int e = 0; e < E_; ++e) acc[e] = 0.f;
  for (int d = lane; d < D_; d += 32) {
    const float xv = xs[(size_t)t * D_ + d];
#pragma unroll
    for (int e = 0; e < E_; ++e) acc[e] += xv * gw[(size_t)e * D_ + d];
  }
#pragma unroll
  for (int e = 0; e < E_; ++e)
    for (int sh = 16; sh > 0; sh >>= 1) acc[e] += __shfl_xor(acc[e], sh, 32);
  if (lane == 0) {
    int i0 = 0; float v0 = acc[0];
    for (int e = 1; e < E_; ++e) if (acc[e] > v0) { v0 = acc[e]; i0 = e; }
    int i1 = (i0 == 0) ? 1 : 0; float v1 = acc[i1];
    for (int e = 0; e < E_; ++e)
      if (e != i0 && acc[e] > v1) { v1 = acc[e]; i1 = e; }
    const float e1 = __expf(v1 - v0);
    const float w0 = 1.f / (1.f + e1);
    const float w1 = e1 / (1.f + e1);
    const int p0 = atomicAdd(&cnts[i0], 1);
    lists[i0 * T_ + p0] = t; wlists[i0 * T_ + p0] = w0;
    const int p1 = atomicAdd(&cnts[i1], 1);
    lists[i1 * T_ + p1] = t; wlists[i1 * T_ + p1] = w1;
  }
}

// ---------------------------------------------------------------------------
// Orchestration
// ---------------------------------------------------------------------------
extern "C" void kernel_launch(void* const* d_in, const int* in_sizes, int n_in,
                              void* d_out, int out_size, void* d_ws, size_t ws_size,
                              hipStream_t stream) {
  const int*   ids      = (const int*)  d_in[0];
  // d_in[1] = positions (arange(S)); implicit in token index
  const float* emb      = (const float*)d_in[2];
  const float* wq       = (const float*)d_in[3];
  const float* wk       = (const float*)d_in[4];
  const float* wv       = (const float*)d_in[5];
  const float* wo       = (const float*)d_in[6];
  const float* attn_nw  = (const float*)d_in[7];
  const float* ffn_nw   = (const float*)d_in[8];
  const float* gate_w   = (const float*)d_in[9];
  const float* w1       = (const float*)d_in[10];
  const float* w2       = (const float*)d_in[11];
  const float* w3       = (const float*)d_in[12];
  const float* final_nw = (const float*)d_in[13];
  const float* out_w    = (const float*)d_in[14];
  float* out = (float*)d_out;

  char* p = (char*)d_ws;
  auto alloc = [&](size_t bytes) -> void* {
    void* r = (void*)p;
    p += (bytes + 255) & ~(size_t)255;
    return r;
  };
  float* h      = (float*)alloc((size_t)T_ * D_ * 4);
  float* x      = (float*)alloc((size_t)T_ * D_ * 4);
  float* qb     = (float*)alloc((size_t)T_ * NH_ * HD_ * 4);
  float* kb     = (float*)alloc((size_t)T_ * NKV_ * HD_ * 4);
  float* vb     = (float*)alloc((size_t)T_ * NKV_ * HD_ * 4);
  float* attn   = (float*)alloc((size_t)T_ * NH_ * HD_ * 4);
  float* G      = (float*)alloc((size_t)T_ * HID_ * 4);
  int*   lists  = (int*)  alloc((size_t)E_ * T_ * 4);
  float* wlists = (float*)alloc((size_t)E_ * T_ * 4);
  int*   cnts   = (int*)  alloc(E_ * 4);

  embed_kernel<<<T_, 256, 0, stream>>>(ids, emb, h);

  for (int l = 0; l < L_; ++l) {
    rmsnorm_kernel<<<T_, 256, 0, stream>>>(h, attn_nw + l * D_, x);

    // Q projection: N=1024 -> LDS-staged multi-wave GEMM
    gemm_lds_kernel<false><<<dim3(T_ / 64, (NH_ * HD_ + 511) / 512), 256, 0, stream>>>(
        x, wq + (size_t)l * NH_ * HD_ * D_, qb, T_, NH_ * HD_, D_);
    // K/V projections: N=256 -> single-wave GEMM
    gemm_wmma_kernel<4, 4, false><<<dim3(T_ / 64, (NKV_ * HD_) / 64), 32, 0, stream>>>(
        x, wk + (size_t)l * NKV_ * HD_ * D_, kb, T_, NKV_ * HD_, D_);
    gemm_wmma_kernel<4, 4, false><<<dim3(T_ / 64, (NKV_ * HD_) / 64), 32, 0, stream>>>(
        x, wv + (size_t)l * NKV_ * HD_ * D_, vb, T_, NKV_ * HD_, D_);

    rope_kernel<<<T_, NH_ * 32, 0, stream>>>(qb, NH_);
    rope_kernel<<<T_, NKV_ * 32, 0, stream>>>(kb, NKV_);

    attention_kernel<<<dim3(S_, NH_, B_), 32, 0, stream>>>(qb, kb, vb, attn);

    // h += attn @ wo^T (accumulator pre-loaded from h)
    gemm_lds_kernel<true><<<dim3(T_ / 64, (D_ + 511) / 512), 256, 0, stream>>>(
        attn, wo + (size_t)l * D_ * NH_ * HD_, h, T_, D_, NH_ * HD_);

    rmsnorm_kernel<<<T_, 256, 0, stream>>>(h, ffn_nw + l * D_, x);

    zero_cnts_kernel<<<1, 32, 0, stream>>>(cnts);
    moe_gate_kernel<<<T_, 32, 0, stream>>>(x, gate_w + (size_t)l * E_ * D_,
                                           cnts, lists, wlists);

    for (int e = 0; e < E_; ++e) {
      moe_gemm13_kernel<<<dim3(T_ / 32, HID_ / 32), 32, 0, stream>>>(
          x, w1 + ((size_t)l * E_ + e) * HID_ * D_,
          w3 + ((size_t)l * E_ + e) * HID_ * D_,
          lists + e * T_, cnts, e, G);
      moe_gemm2_kernel<<<dim3(T_ / 32, D_ / 64), 32, 0, stream>>>(
          G, w2 + ((size_t)l * E_ + e) * D_ * HID_,
          lists + e * T_, wlists + e * T_, cnts, e, h);
    }
  }

  rmsnorm_kernel<<<T_, 256, 0, stream>>>(h, final_nw, x);
  gemm_lds_kernel<false><<<dim3(T_ / 64, (V_ + 511) / 512), 256, 0, stream>>>(
      x, out_w, out, T_, V_, D_);
}
